// GAT_62173946577598
// MI455X (gfx1250) — matrix operand order repaired and checked by reading, not compiled, not run on hardware
//
#include <hip/hip_runtime.h>
#include <hip/hip_bf16.h>
#include <math.h>

// ---------------------------------------------------------------------------
// GAT (2x GATConv + linear head) for gfx1250.
// GEMMs: v_wmma_f32_16x16x32_f16 (f16 multiplicands, f32 accum), operands
// staged into LDS by the Tensor Data Mover (tensor_load_to_lds + TENSORcnt).
// Edge softmax/aggregation uses float atomics (L2-resident working set).
// ---------------------------------------------------------------------------

typedef __attribute__((ext_vector_type(16))) _Float16     v16h;
typedef __attribute__((ext_vector_type(8)))  float        v8f;
typedef __attribute__((ext_vector_type(4)))  unsigned int u32x4;
typedef __attribute__((ext_vector_type(8)))  unsigned int u32x8;

#define NEG_SLOPE 0.2f
#define FEAT 128
#define HEADS 4
#define CPH 32          // channels per head
#define NCLS 40

// ---------------- utility kernels ----------------

__global__ void gat_fill_kernel(float* __restrict__ p, float v, long long n) {
    long long i = (long long)blockIdx.x * blockDim.x + threadIdx.x;
    if (i < n) p[i] = v;
}

__global__ void gat_bias_relu_kernel(float* __restrict__ h,
                                     const float* __restrict__ b, long long n) {
    long long i = (long long)blockIdx.x * blockDim.x + threadIdx.x;
    if (i < n) {
        float v = h[i] + b[i & (FEAT - 1)];
        h[i] = v > 0.0f ? v : 0.0f;
    }
}

// ---------------- TDM descriptor helpers (ISA 08_async_tensor §8) ----------
// Group 0: [1:0]=count=1, [63:32]=lds_addr, [120:64]=global_addr, [127:126]=2
// Group 1: [17:16]=data_size(2=4B), [79:48]=tensor_dim0, [111:80]=tensor_dim1,
//          [127:112]=tile_dim0, [143:128]=tile_dim1, [159:144]=tile_dim2,
//          [207:160]=tensor_dim0_stride

__device__ __forceinline__ u32x4 gat_tdm_g0(unsigned lds_addr,
                                            unsigned long long gaddr) {
    u32x4 g;
    g.x = 1u;                                   // count=1, user mode
    g.y = lds_addr;                             // LDS byte address
    g.z = (unsigned)(gaddr & 0xFFFFFFFFull);    // global addr [31:0]
    g.w = (unsigned)((gaddr >> 32) & 0x1FFFFFFu) | (2u << 30); // [56:32]|type=2
    return g;
}

__device__ __forceinline__ u32x8 gat_tdm_g1(unsigned td0, unsigned td1,
                                            unsigned tile0, unsigned tile1,
                                            unsigned stride0) {
    u32x8 g;
    g.s0 = 2u << 16;                            // data_size = 4 bytes
    g.s1 = (td0 & 0xFFFFu) << 16;               // tensor_dim0[15:0]
    g.s2 = (td0 >> 16) | ((td1 & 0xFFFFu) << 16);
    g.s3 = (td1 >> 16) | ((tile0 & 0xFFFFu) << 16);
    g.s4 = (tile1 & 0xFFFFu);                   // tile_dim1, tile_dim2=0
    g.s5 = stride0;                             // tensor_dim0_stride[31:0]
    g.s6 = 0u;
    g.s7 = 0u;
    return g;
}

__device__ __forceinline__ void gat_tdm_load(u32x4 g0, u32x8 g1) {
    asm volatile("tensor_load_to_lds %0, %1" :: "s"(g0), "s"(g1) : "memory");
}

// ---------------- WMMA GEMM: Y[N,128] = X[N,128] * W[128,128] ----------------
// One block = 16 output rows, 8 waves each owning a 16-col tile.
// TDM stages raw f32 W (1-D tile) and the 16x128 X tile (2-D tile) into LDS;
// a convert pass produces f16 operands (W transposed so B frags are
// contiguous in K).

__global__ __launch_bounds__(256) void gat_gemm128_wmma(
    const float* __restrict__ X, const float* __restrict__ W,
    float* __restrict__ Y, int N)
{
    __shared__ float    ldsWf32[FEAT * FEAT];   // raw W, 64KB (TDM dest)
    __shared__ float    ldsXf32[16 * FEAT];     // raw X tile, 8KB (TDM dest)
    __shared__ _Float16 ldsWt[FEAT * FEAT];     // [col][k] 32KB
    __shared__ _Float16 ldsX[16 * FEAT];        // [row][k]  4KB

    const int t    = threadIdx.x;
    const int wave = t >> 5;
    const int lane = t & 31;
    const int rowbase = blockIdx.x * 16;

    if (wave == 0) {
        // W: contiguous 16384-element 1-D tile
        gat_tdm_load(
            gat_tdm_g0((unsigned)(size_t)&ldsWf32[0],
                       (unsigned long long)(size_t)W),
            gat_tdm_g1(/*td0=*/FEAT * FEAT, /*td1=*/1,
                       /*tile0=*/FEAT * FEAT, /*tile1=*/1,
                       /*stride0=*/FEAT * FEAT));
        // X tile: 128 x rows_left 2-D tile, row stride 128 elements
        unsigned rows_left = (unsigned)((N - rowbase) < 16 ? (N - rowbase) : 16);
        gat_tdm_load(
            gat_tdm_g0((unsigned)(size_t)&ldsXf32[0],
                       (unsigned long long)(size_t)(X + (size_t)rowbase * FEAT)),
            gat_tdm_g1(/*td0=*/FEAT, /*td1=*/rows_left,
                       /*tile0=*/FEAT, /*tile1=*/16,
                       /*stride0=*/FEAT));
        __builtin_amdgcn_s_wait_tensorcnt(0);
    }
    __syncthreads();

    // convert: W f32 -> f16 transposed, X f32 -> f16
    for (int idx = t; idx < FEAT * FEAT; idx += 256) {
        int k = idx >> 7, n = idx & (FEAT - 1);
        ldsWt[n * FEAT + k] = (_Float16)ldsWf32[idx];
    }
    for (int idx = t; idx < 16 * FEAT; idx += 256) {
        ldsX[idx] = (_Float16)ldsXf32[idx];
    }
    __syncthreads();

    const int colbase = wave * 16;
    const int lr = lane & 15;    // row (A) / col (B,D) within tile
    const int hi = lane >> 4;    // K-half selector
    v8f acc = {};

#pragma unroll
    for (int kt = 0; kt < 4; ++kt) {
        const int k0 = kt * 32;
        v16h a, b;
        // A: 16x32 f16. lane<16: K=k0+0..15 ; lane>=16: K=k0+16..31
        const _Float16* xr = &ldsX[lr * FEAT + k0 + hi * 16];
#pragma unroll
        for (int i = 0; i < 8; ++i) { a[i] = xr[i]; a[8 + i] = xr[8 + i]; }
        // B: 32x16 f16. lane<16: K=k0+0..15 ; lane>=16: K=k0+16..31
        const _Float16* wc = &ldsWt[(colbase + lr) * FEAT + k0 + hi * 16];
#pragma unroll
        for (int j = 0; j < 16; ++j) b[j] = wc[j];
        acc = __builtin_amdgcn_wmma_f32_16x16x32_f16(
            false, a, false, b, (short)0, acc, false, false);
    }

    // D: VGPR v -> row = hi*8 + v, col = lr
    if (rowbase + 16 <= N) {                       // uniform fast path
        float* yp = &Y[(size_t)(rowbase + hi * 8) * FEAT + colbase + lr];
#pragma unroll
        for (int v = 0; v < 8; ++v) yp[(size_t)v * FEAT] = acc[v];
    } else {
#pragma unroll
        for (int v = 0; v < 8; ++v) {
            int row = rowbase + hi * 8 + v;
            if (row < N) Y[(size_t)row * FEAT + colbase + lr] = acc[v];
        }
    }
}

// ---------------- classifier GEMM: Y[N,40] = X[N,128]*Wc[128,40] + bc --------

__global__ __launch_bounds__(128) void gat_gemm_cls_wmma(
    const float* __restrict__ X, const float* __restrict__ Wc,
    const float* __restrict__ bc, float* __restrict__ Y, int N)
{
    __shared__ _Float16 ldsWt[48 * FEAT];   // cols padded 40 -> 48, 12KB
    __shared__ _Float16 ldsX[16 * FEAT];

    const int t    = threadIdx.x;
    const int wave = t >> 5;
    const int lane = t & 31;
    const int rowbase = blockIdx.x * 16;

    for (int idx = t; idx < 48 * FEAT; idx += 128) {
        int n = idx >> 7, k = idx & (FEAT - 1);
        ldsWt[idx] = (n < NCLS) ? (_Float16)Wc[k * NCLS + n] : (_Float16)0.0f;
    }
    for (int idx = t; idx < 16 * FEAT; idx += 128) {
        int r = idx >> 7, k = idx & (FEAT - 1);
        int rr = rowbase + r; if (rr >= N) rr = N - 1;
        ldsX[idx] = (_Float16)X[(size_t)rr * FEAT + k];
    }
    __syncthreads();

    if (wave < 3) {
        const int colbase = wave * 16;
        const int lr = lane & 15;
        const int hi = lane >> 4;
        v8f acc = {};
#pragma unroll
        for (int kt = 0; kt < 4; ++kt) {
            const int k0 = kt * 32;
            v16h a, b;
            const _Float16* xr = &ldsX[lr * FEAT + k0 + hi * 16];
#pragma unroll
            for (int i = 0; i < 8; ++i) { a[i] = xr[i]; a[8 + i] = xr[8 + i]; }
            const _Float16* wc = &ldsWt[(colbase + lr) * FEAT + k0 + hi * 16];
#pragma unroll
            for (int j = 0; j < 16; ++j) b[j] = wc[j];
            acc = __builtin_amdgcn_wmma_f32_16x16x32_f16(
                false, a, false, b, (short)0, acc, false, false);
        }
        const int col = colbase + lr;
        const bool colok = (col < NCLS);
        const float bias = colok ? bc[col] : 0.0f;
        if (rowbase + 16 <= N) {
            if (colok) {
                float* yp = &Y[(size_t)(rowbase + hi * 8) * NCLS + col];
#pragma unroll
                for (int v = 0; v < 8; ++v) yp[(size_t)v * NCLS] = acc[v] + bias;
            }
        } else {
#pragma unroll
            for (int v = 0; v < 8; ++v) {
                int row = rowbase + hi * 8 + v;
                if (row < N && colok)
                    Y[(size_t)row * NCLS + col] = acc[v] + bias;
            }
        }
    }
}

// ---------------- per-node attention coefficients ----------------

__global__ void gat_attn_kernel(const float* __restrict__ H,
                                const float* __restrict__ att_src,
                                const float* __restrict__ att_dst,
                                float* __restrict__ asrc,
                                float* __restrict__ adst, int N)
{
    int i = blockIdx.x * blockDim.x + threadIdx.x;  // node*HEADS + head
    if (i >= N * HEADS) return;
    const int n = i >> 2, h = i & 3;
    const float* hp = H + (size_t)n * FEAT + h * CPH;
    const float* sv = att_src + h * CPH;
    const float* dv = att_dst + h * CPH;
    float aS = 0.0f, aD = 0.0f;
#pragma unroll
    for (int c = 0; c < CPH; ++c) {
        float v = hp[c];
        aS += v * sv[c];
        aD += v * dv[c];
    }
    asrc[i] = aS;
    adst[i] = aD;
}

// ---------------- edge softmax (max / denom / scatter) ----------------

__device__ __forceinline__ float gat_atomic_max_f32(float* addr, float value) {
    // sign-split trick: valid given init = -inf
    if (value >= 0.0f)
        return __int_as_float(atomicMax((int*)addr, __float_as_int(value)));
    else
        return __uint_as_float(atomicMin((unsigned int*)addr, __float_as_uint(value)));
}

__device__ __forceinline__ float gat_leaky(float e) {
    return e > 0.0f ? e : NEG_SLOPE * e;
}

__global__ void gat_edge_max_kernel(const int* __restrict__ src,
                                    const int* __restrict__ dst,
                                    const float* __restrict__ asrc,
                                    const float* __restrict__ adst,
                                    float* __restrict__ emax, int E, int ETOT)
{
    long long tid = (long long)blockIdx.x * blockDim.x + threadIdx.x;
    int i = (int)(tid >> 2), h = (int)(tid & 3);
    if (i >= ETOT) return;
    int s = (i < E) ? src[i] : (i - E);
    int d = (i < E) ? dst[i] : (i - E);
    float e = gat_leaky(asrc[s * HEADS + h] + adst[d * HEADS + h]);
    gat_atomic_max_f32(&emax[d * HEADS + h], e);
}

__global__ void gat_edge_denom_kernel(const int* __restrict__ src,
                                      const int* __restrict__ dst,
                                      const float* __restrict__ asrc,
                                      const float* __restrict__ adst,
                                      const float* __restrict__ emax,
                                      float* __restrict__ denom, int E, int ETOT)
{
    long long tid = (long long)blockIdx.x * blockDim.x + threadIdx.x;
    int i = (int)(tid >> 2), h = (int)(tid & 3);
    if (i >= ETOT) return;
    int s = (i < E) ? src[i] : (i - E);
    int d = (i < E) ? dst[i] : (i - E);
    float e = gat_leaky(asrc[s * HEADS + h] + adst[d * HEADS + h]);
    atomicAdd(&denom[d * HEADS + h], __expf(e - emax[d * HEADS + h]));
}

__global__ void gat_edge_scatter_kernel(const int* __restrict__ src,
                                        const int* __restrict__ dst,
                                        const float* __restrict__ asrc,
                                        const float* __restrict__ adst,
                                        const float* __restrict__ emax,
                                        const float* __restrict__ denom,
                                        const float* __restrict__ H,
                                        float* __restrict__ agg, int E, int ETOT)
{
    long long tid = (long long)blockIdx.x * blockDim.x + threadIdx.x;
    int i = (int)(tid >> 7);              // edge id
    int f = (int)(tid & (FEAT - 1));      // feature channel
    if (i >= ETOT) return;
    int h = f >> 5;                        // head
    int s = (i < E) ? src[i] : (i - E);
    int d = (i < E) ? dst[i] : (i - E);
    float e = gat_leaky(asrc[s * HEADS + h] + adst[d * HEADS + h]);
    float alpha = __expf(e - emax[d * HEADS + h]) /
                  (denom[d * HEADS + h] + 1e-16f);
    atomicAdd(&agg[(size_t)d * FEAT + f], alpha * H[(size_t)s * FEAT + f]);
}

// ---------------- host orchestration ----------------

static inline unsigned gat_blocks(long long n, int bs) {
    return (unsigned)((n + bs - 1) / bs);
}

extern "C" void kernel_launch(void* const* d_in, const int* in_sizes, int n_in,
                              void* d_out, int out_size, void* d_ws, size_t ws_size,
                              hipStream_t stream) {
    (void)n_in; (void)out_size; (void)ws_size;

    const float* x        = (const float*)d_in[0];
    const int*   eidx     = (const int*)d_in[1];
    const float* W1       = (const float*)d_in[2];
    const float* att_src1 = (const float*)d_in[3];
    const float* att_dst1 = (const float*)d_in[4];
    const float* b1       = (const float*)d_in[5];
    const float* W2       = (const float*)d_in[6];
    const float* att_src2 = (const float*)d_in[7];
    const float* att_dst2 = (const float*)d_in[8];
    const float* b2       = (const float*)d_in[9];
    const float* Wc       = (const float*)d_in[10];
    const float* bc       = (const float*)d_in[11];
    float* out = (float*)d_out;

    const int N = in_sizes[0] / FEAT;
    const int E = in_sizes[1] / 2;
    const int ETOT = E + N;                 // + self loops
    const int* src = eidx;
    const int* dst = eidx + E;

    // workspace layout (floats)
    float* H     = (float*)d_ws;                    // [N,128]
    float* AGG   = H     + (size_t)N * FEAT;        // [N,128]
    float* asrc  = AGG   + (size_t)N * FEAT;        // [N,4]
    float* adst  = asrc  + (size_t)N * HEADS;
    float* emax  = adst  + (size_t)N * HEADS;
    float* denom = emax  + (size_t)N * HEADS;

    const int rowTiles = (N + 15) / 16;
    const long long nNH   = (long long)N * HEADS;
    const long long nNF   = (long long)N * FEAT;
    const long long nEH   = (long long)ETOT * HEADS;
    const long long nEF   = (long long)ETOT * FEAT;

    const float* layer_in = x;
    for (int layer = 0; layer < 2; ++layer) {
        const float* W  = (layer == 0) ? W1 : W2;
        const float* as = (layer == 0) ? att_src1 : att_src2;
        const float* ad = (layer == 0) ? att_dst1 : att_dst2;
        const float* bb = (layer == 0) ? b1 : b2;

        // H = layer_in @ W   (TDM-staged WMMA)
        gat_gemm128_wmma<<<rowTiles, 256, 0, stream>>>(layer_in, W, H, N);
        // attention coefficients
        gat_attn_kernel<<<gat_blocks(nNH, 256), 256, 0, stream>>>(H, as, ad, asrc, adst, N);
        // init softmax state + aggregation buffer (AGG was layer_in for layer 2,
        // but the GEMM above has fully consumed it before this point on-stream)
        gat_fill_kernel<<<gat_blocks(nNH, 256), 256, 0, stream>>>(emax, -INFINITY, nNH);
        gat_fill_kernel<<<gat_blocks(nNH, 256), 256, 0, stream>>>(denom, 0.0f, nNH);
        gat_fill_kernel<<<gat_blocks(nNF, 256), 256, 0, stream>>>(AGG, 0.0f, nNF);
        // segment softmax over incoming edges
        gat_edge_max_kernel<<<gat_blocks(nEH, 256), 256, 0, stream>>>(
            src, dst, asrc, adst, emax, E, ETOT);
        gat_edge_denom_kernel<<<gat_blocks(nEH, 256), 256, 0, stream>>>(
            src, dst, asrc, adst, emax, denom, E, ETOT);
        gat_edge_scatter_kernel<<<gat_blocks(nEF, 256), 256, 0, stream>>>(
            src, dst, asrc, adst, emax, denom, H, AGG, E, ETOT);
        // bias + relu (in place); AGG becomes next layer's input
        gat_bias_relu_kernel<<<gat_blocks(nNF, 256), 256, 0, stream>>>(AGG, bb, nNF);
        layer_in = AGG;
    }

    // classifier head (WMMA, padded 40->48 cols, bias fused)
    gat_gemm_cls_wmma<<<rowTiles, 128, 0, stream>>>(AGG, Wc, bc, out, N);
}
